// EBLNN_11991548690793
// MI455X (gfx1250) — matrix operand order repaired and checked by reading, not compiled, not run on hardware
//
#include <hip/hip_runtime.h>

// ============================================================================
// Persistent CfC/LSTM recurrent kernel for MI455X (gfx1250, wave32, WMMA).
//   - 64 persistent workgroups x 256 threads (1 WG per WGP), 3 device-wide
//     software barriers per timestep
//   - all matmuls via v_wmma_f32_16x16x32_bf16 (f32 accumulate)
//   - loop-invariant weight slices staged once into 320 KB LDS (full WGP
//     capacity), pre-swizzled into the WMMA B-operand register layout;
//     Wh slice staged via TDM tensor_load_to_lds (s_wait_tensorcnt)
//   - inner loops preload all 4 B tiles per k-step to hide LDS latency
// ============================================================================

#define DEVINL __device__ __forceinline__

typedef __attribute__((ext_vector_type(16))) __bf16         v16bf;
typedef __attribute__((ext_vector_type(8)))  float          v8f;
typedef __attribute__((ext_vector_type(8)))  unsigned short u16x8;
typedef __attribute__((ext_vector_type(16))) unsigned short u16x16;
typedef __attribute__((ext_vector_type(4)))  unsigned int   u32x4;
typedef __attribute__((ext_vector_type(8)))  int            i32x8;
typedef __attribute__((ext_vector_type(4)))  int            i32x4;

constexpr int B   = 256;
constexpr int T   = 512;
constexpr int F   = 5;
constexpr int H   = 512;
constexpr int BBU = 128;

constexpr int NBLK = 64;   // persistent workgroups (must be co-resident)
constexpr int NTHR = 256;  // 8 waves of 32

// LDS: Wh slice 256KB + ff/ta/tb slices 64KB = 320KB (full WGP LDS)
constexpr size_t LDS_WH_BYTES = (size_t)4 * 64 * H * 2;        // 262144
constexpr size_t LDS_FF_BYTES = (size_t)4 * 64 * BBU * 2;      // 65536
constexpr size_t LDS_BYTES    = LDS_WH_BYTES + LDS_FF_BYTES;   // 327680

// ---------------- workspace layout (bytes, 256-aligned chunks) --------------
constexpr size_t OFF_BAR  = 0;                                   // 2 x u32
constexpr size_t OFF_HBF  = 256;                                 // h      bf16 [B,H]
constexpr size_t OFF_C    = OFF_HBF  + (size_t)B * H * 2;        // c      f32  [B,H]
constexpr size_t OFF_HL   = OFF_C    + (size_t)B * H * 4;        // h_lstm bf16 [B,H]
constexpr size_t OFF_FEAT = OFF_HL   + (size_t)B * H * 2;        // feat   bf16 [B,BB]
constexpr size_t OFF_HF   = OFF_FEAT + (size_t)B * BBU * 2;      // h      f32  [B,H]
constexpr size_t OFF_WHB  = OFF_HF   + (size_t)B * H * 4;        // Wh     bf16 [4H,H]
constexpr size_t OFF_BBB  = OFF_WHB  + (size_t)4 * H * H * 2;    // bb_W   bf16 [BB,F+H]
constexpr size_t OFF_FF1  = OFF_BBB  + (size_t)BBU * (F + H) * 2;
constexpr size_t OFF_FF2  = OFF_FF1  + (size_t)H * BBU * 2;
constexpr size_t OFF_TA   = OFF_FF2  + (size_t)H * BBU * 2;
constexpr size_t OFF_TB   = OFF_TA   + (size_t)H * BBU * 2;

// ---------------- output layout (floats) ------------------------------------
constexpr size_t OUT_Y  = 0;                          // y_pred [B,T,2]
constexpr size_t OUT_E  = (size_t)B * T * 2;          // e_pred [B,T,1]
constexpr size_t OUT_HL = OUT_E + (size_t)B * T;      // h_last [B,H]
constexpr size_t OUT_CL = OUT_HL + (size_t)B * H;     // c_last [B,H]

// ---------------- helpers ----------------------------------------------------
DEVINL unsigned short f2bf(float f) {
  union { float f; unsigned u; } v; v.f = f;
  unsigned u = v.u;
  u += 0x7FFFu + ((u >> 16) & 1u);     // round-to-nearest-even
  return (unsigned short)(u >> 16);
}

DEVINL float sigmoidf(float x) { return 1.0f / (1.0f + __expf(-x)); }

DEVINL v8f zero8() {
  v8f z;
#pragma unroll
  for (int i = 0; i < 8; ++i) z[i] = 0.0f;
  return z;
}

// A-matrix 16x32 bf16, contiguous-k order (k = 16*half + s, same as B): lane m (0..15) rows; lane-half splits K.
DEVINL v16bf load_a16(const unsigned short* __restrict__ base, int ld,
                      int m0, int k0, int lane) {
  const unsigned short* p = base + (size_t)(m0 + (lane & 15)) * ld + k0 + 16 * (lane >> 4);
  u16x8 lo = *(const u16x8*)(p);
  u16x8 hi = *(const u16x8*)(p + 8);
  u16x16 v;
#pragma unroll
  for (int i = 0; i < 8; ++i) { v[i] = lo[i]; v[8 + i] = hi[i]; }
  return __builtin_bit_cast(v16bf, v);
}

// B-matrix 32x16 bf16 from global (W[col][k] contiguous in k)
DEVINL v16bf load_b16(const unsigned short* __restrict__ base, int ld,
                      int n0, int k0, int lane) {
  const unsigned short* p = base + (size_t)(n0 + (lane & 15)) * ld + k0 + 16 * (lane >> 4);
  u16x16 v = *(const u16x16*)(p);
  return __builtin_bit_cast(v16bf, v);
}

// same but with an element offset that breaks 32B alignment (bb_W, ld=517)
DEVINL v16bf load_b16_off(const unsigned short* __restrict__ base, int ld,
                          int n0, int eoff, int k0, int lane) {
  const unsigned short* p = base + (size_t)(n0 + (lane & 15)) * ld + eoff + k0 + 16 * (lane >> 4);
  u16x16 v;
#pragma unroll
  for (int i = 0; i < 16; ++i) v[i] = p[i];
  return __builtin_bit_cast(v16bf, v);
}

// B-matrix tile from LDS: tiles stored pre-swizzled, lane's 16 halves contiguous.
DEVINL v16bf load_b16_lds(const unsigned char* sm, int blk, int lane) {
  const u16x16* p = (const u16x16*)(sm + (size_t)blk * 1024 + lane * 32);
  return __builtin_bit_cast(v16bf, *p);
}

// device-wide barrier: agent-scope arrival counter + generation (sense)
DEVINL void gbar(unsigned* cnt, unsigned* gen) {
  __syncthreads();
  if (threadIdx.x == 0) {
    unsigned g = __hip_atomic_load(gen, __ATOMIC_RELAXED, __HIP_MEMORY_SCOPE_AGENT);
    unsigned a = __hip_atomic_fetch_add(cnt, 1u, __ATOMIC_ACQ_REL, __HIP_MEMORY_SCOPE_AGENT);
    if (a == (unsigned)(NBLK - 1)) {
      __hip_atomic_store(cnt, 0u, __ATOMIC_RELAXED, __HIP_MEMORY_SCOPE_AGENT);
      __hip_atomic_fetch_add(gen, 1u, __ATOMIC_ACQ_REL, __HIP_MEMORY_SCOPE_AGENT);
    } else {
      while (__hip_atomic_load(gen, __ATOMIC_ACQUIRE, __HIP_MEMORY_SCOPE_AGENT) == g) {
        __builtin_amdgcn_s_sleep(2);
      }
    }
  }
  __syncthreads();
}

// ============================================================================
// init: zero state + barrier, convert weights fp32 -> bf16
// ============================================================================
__global__ void eblnn_init(const float* __restrict__ Wh, const float* __restrict__ bbW,
                           const float* __restrict__ W1, const float* __restrict__ W2,
                           const float* __restrict__ W3, const float* __restrict__ W4,
                           unsigned char* __restrict__ ws) {
  int idx = blockIdx.x * blockDim.x + threadIdx.x;
  int stride = gridDim.x * blockDim.x;
  if (idx == 0) {
    ((unsigned*)(ws + OFF_BAR))[0] = 0u;
    ((unsigned*)(ws + OFF_BAR))[1] = 0u;
  }
  unsigned short* hbf = (unsigned short*)(ws + OFF_HBF);
  float*          cst = (float*)(ws + OFF_C);
  for (int i = idx; i < B * H; i += stride) { hbf[i] = 0; cst[i] = 0.0f; }

  unsigned short* whb = (unsigned short*)(ws + OFF_WHB);
  for (int i = idx; i < 4 * H * H; i += stride) whb[i] = f2bf(Wh[i]);

  unsigned short* bbb = (unsigned short*)(ws + OFF_BBB);
  for (int i = idx; i < BBU * (F + H); i += stride) bbb[i] = f2bf(bbW[i]);

  unsigned short* p1 = (unsigned short*)(ws + OFF_FF1);
  unsigned short* p2 = (unsigned short*)(ws + OFF_FF2);
  unsigned short* p3 = (unsigned short*)(ws + OFF_TA);
  unsigned short* p4 = (unsigned short*)(ws + OFF_TB);
  for (int i = idx; i < H * BBU; i += stride) {
    p1[i] = f2bf(W1[i]); p2[i] = f2bf(W2[i]);
    p3[i] = f2bf(W3[i]); p4[i] = f2bf(W4[i]);
  }
}

// ============================================================================
// persistent recurrent kernel
// ============================================================================
__global__ void __launch_bounds__(NTHR, 1)
eblnn_persistent(const float* __restrict__ x,
                 const float* __restrict__ lstm_Wi, const float* __restrict__ lstm_bi,
                 const float* __restrict__ bb_Wf,   const float* __restrict__ bb_b,
                 const float* __restrict__ ff1_bv,  const float* __restrict__ ff2_bv,
                 const float* __restrict__ ta_bv,   const float* __restrict__ tb_bv,
                 const float* __restrict__ pred_W,  const float* __restrict__ pred_b,
                 const float* __restrict__ en_W,    const float* __restrict__ en_b,
                 unsigned char* __restrict__ ws,    float* __restrict__ out) {
  extern __shared__ __align__(32) unsigned char smem[];
  unsigned char* ldsWh = smem;                   // 256 KB: Wh B-tiles, swizzled
  unsigned char* ldsFF = smem + LDS_WH_BYTES;    //  64 KB: ff1/ff2/ta/tb B-tiles

  unsigned* bar_cnt = (unsigned*)(ws + OFF_BAR);
  unsigned* bar_gen = bar_cnt + 1;

  unsigned short* h_bf  = (unsigned short*)(ws + OFF_HBF);
  float*          c_st  = (float*)(ws + OFF_C);
  unsigned short* hl_bf = (unsigned short*)(ws + OFF_HL);
  unsigned short* ft_bf = (unsigned short*)(ws + OFF_FEAT);
  float*          h_f32 = (float*)(ws + OFF_HF);
  const unsigned short* Whb = (const unsigned short*)(ws + OFF_WHB);
  const unsigned short* bbb = (const unsigned short*)(ws + OFF_BBB);
  const unsigned short* wf1 = (const unsigned short*)(ws + OFF_FF1);
  const unsigned short* wf2 = (const unsigned short*)(ws + OFF_FF2);
  const unsigned short* wta = (const unsigned short*)(ws + OFF_TA);
  const unsigned short* wtb = (const unsigned short*)(ws + OFF_TB);

  const int w    = blockIdx.x;            // 0..63
  const int wave = threadIdx.x >> 5;      // 0..7
  const int lane = threadIdx.x & 31;
  const int l15  = lane & 15;
  const int lhi  = lane >> 4;

  // static partition: stages 1/3 -> (32 batch rows) x (64 cols, all 4 gates)
  const int rb0  = (w & 7) * 32;          // batch-row base
  const int jg0  = (w >> 3) * 64;         // col-group base (h / gate space)
  const int ct16 = (w >> 3) * 16;         // stage-2 feat-col tile
  const int rt = wave >> 2, ct = wave & 3;
  const int m0 = rb0 + 16 * rt;           // stage 1/3 row tile
  const int n0 = jg0 + 16 * ct;           // stage 1/3 col tile

  // ---- one-time LDS staging of loop-invariant weight slices ----
  // Wh slice: blocks (g,c,kc) = 4*4*16 x 1KB; LDS linear order inside a
  // (g,c) pair is exactly a 4-D regular tile walk of the global layout:
  //   dim0: 32B contiguous (4 x 8B elements)
  //   dim1: 16 weight cols,   stride 1024B (128 elem)
  //   dim2: 2 lane-half Ks,   stride   32B (  4 elem)
  //   dim3: 16 K-chunks,      stride   64B (  8 elem)
#if __has_builtin(__builtin_amdgcn_tensor_load_to_lds) && __has_builtin(__builtin_amdgcn_s_wait_tensorcnt)
  if (wave == 0) {
    unsigned long long gbase = (unsigned long long)(const void*)Whb;
    unsigned ldsbase = (unsigned)(unsigned long long)(void*)ldsWh;
#pragma unroll 1
    for (int g = 0; g < 4; ++g) {
#pragma unroll 1
      for (int c = 0; c < 4; ++c) {
        unsigned long long ga =
            gbase + (unsigned long long)(g * H + jg0 + c * 16) * (size_t)(H * 2);
        unsigned la = ldsbase + (unsigned)((g * 4 + c) * 16) * 1024u;
        u32x4 d0; i32x8 d1; i32x4 d2, d3;
        d0[0] = 1u;                                  // count=1, user descriptor
        d0[1] = la;                                  // lds_addr
        d0[2] = (unsigned)ga;                        // global_addr[31:0]
        d0[3] = (unsigned)((ga >> 32) & 0x01FFFFFFu) // global_addr[56:32]
              | 0x80000000u;                         // type=2 ("image")
        d1[0] = 0x00030000;                          // data_size=3 (8B)
        d1[1] = 4 << 16;                             // tensor_dim0 = 4
        d1[2] = 16 << 16;                            // tensor_dim1 = 16
        d1[3] = 4 << 16;                             // tile_dim0 = 4
        d1[4] = 16 | (2 << 16);                      // tile_dim1=16, tile_dim2=2
        d1[5] = 128;                                 // tensor_dim0_stride = 128
        d1[6] = 4 << 16;                             // tensor_dim1_stride = 4
        d1[7] = 0;
        d2[0] = 2;                                   // tensor_dim2 = 2
        d2[1] = 16;                                  // tensor_dim3 = 16
        d2[2] = 8;                                   // tensor_dim2_stride = 8
        d2[3] = 16 << 16;                            // tile_dim3 = 16
        d3[0] = 0; d3[1] = 0; d3[2] = 0; d3[3] = 0;  // no dim4
#if defined(__clang_major__) && (__clang_major__ >= 23)
        i32x8 d4 = {0, 0, 0, 0, 0, 0, 0, 0};
        __builtin_amdgcn_tensor_load_to_lds(d0, d1, d2, d3, d4, 0);
#else
        __builtin_amdgcn_tensor_load_to_lds(d0, d1, d2, d3, 0);
#endif
      }
    }
    __builtin_amdgcn_s_wait_tensorcnt(0);
  }
#else
  for (int blk = wave; blk < 256; blk += 8) {
    int g  = blk >> 6;
    int c  = (blk >> 4) & 3;
    int kc = blk & 15;
    const unsigned short* p =
        Whb + (size_t)(g * H + jg0 + c * 16 + l15) * H + kc * 32 + 16 * lhi;
    u16x16 v = *(const u16x16*)(p);
    *(u16x16*)(ldsWh + (size_t)blk * 1024 + lane * 32) = v;
  }
#endif
  {
    const unsigned short* fw[4] = {wf1, wf2, wta, wtb};
    for (int blk = wave; blk < 64; blk += 8) {
      int mat = blk >> 4;
      int c   = (blk >> 2) & 3;
      int kc  = blk & 3;
      const unsigned short* p =
          fw[mat] + (size_t)(jg0 + c * 16 + l15) * BBU + kc * 32 + 16 * lhi;
      u16x16 v = *(const u16x16*)(p);
      *(u16x16*)(ldsFF + (size_t)blk * 1024 + lane * 32) = v;
    }
  }
  __syncthreads();

  // ---- hoisted per-lane constants (loop-invariant in t) ----
  const int jlane = n0 + l15;             // this lane's output column (0..511)
  float wi_l[4][F], bi_l[4];
#pragma unroll
  for (int g = 0; g < 4; ++g) {
    int col = g * H + jlane;
    bi_l[g] = lstm_bi[col];
#pragma unroll
    for (int f = 0; f < F; ++f) wi_l[g][f] = lstm_Wi[col * F + f];
  }
  const float b_f1 = ff1_bv[jlane], b_f2 = ff2_bv[jlane];
  const float b_ta = ta_bv[jlane],  b_tb = tb_bv[jlane];

  const int clane = ct16 + l15;           // stage-2 feat column (0..127)
  const float bbbias = bb_b[clane];
  float bbx[F];
#pragma unroll
  for (int f = 0; f < F; ++f) bbx[f] = bb_Wf[clane * (F + H) + f];

#pragma unroll 1
  for (int t = 0; t < T; ++t) {
    // ================= stage 1: LSTM  z = x@Wi.T + bi + h@Wh.T =============
    {
      v8f acc[4];
#pragma unroll
      for (int g = 0; g < 4; ++g) acc[g] = zero8();
#pragma unroll 4
      for (int k0 = 0; k0 < H; k0 += 32) {
        int kc = k0 >> 5;
        v16bf a  = load_a16(h_bf, H, m0, k0, lane);   // recurrent state (L2)
        // preload all 4 gate B tiles -> staggered s_wait_dscnt, hides LDS lat
        v16bf bm0 = load_b16_lds(ldsWh, (0 * 4 + ct) * 16 + kc, lane);
        v16bf bm1 = load_b16_lds(ldsWh, (1 * 4 + ct) * 16 + kc, lane);
        v16bf bm2 = load_b16_lds(ldsWh, (2 * 4 + ct) * 16 + kc, lane);
        v16bf bm3 = load_b16_lds(ldsWh, (3 * 4 + ct) * 16 + kc, lane);
        acc[0] = __builtin_amdgcn_wmma_f32_16x16x32_bf16(
            false, a, false, bm0, (short)0, acc[0], false, false);
        acc[1] = __builtin_amdgcn_wmma_f32_16x16x32_bf16(
            false, a, false, bm1, (short)0, acc[1], false, false);
        acc[2] = __builtin_amdgcn_wmma_f32_16x16x32_bf16(
            false, a, false, bm2, (short)0, acc[2], false, false);
        acc[3] = __builtin_amdgcn_wmma_f32_16x16x32_bf16(
            false, a, false, bm3, (short)0, acc[3], false, false);
      }
#pragma unroll
      for (int r = 0; r < 8; ++r) {
        int b = m0 + r + 8 * lhi;
        const float* xp = x + ((size_t)b * T + t) * F;
        float xv[F];
#pragma unroll
        for (int f = 0; f < F; ++f) xv[f] = xp[f];
        float z[4];
#pragma unroll
        for (int g = 0; g < 4; ++g) {
          float zz = acc[g][r] + bi_l[g];
#pragma unroll
          for (int f = 0; f < F; ++f) zz += xv[f] * wi_l[g][f];
          z[g] = zz;
        }
        int idx = b * H + jlane;
        float c_old = c_st[idx];
        float cn = c_old * sigmoidf(z[2] + 1.0f) + tanhf(z[0]) * sigmoidf(z[1]);
        float hl = tanhf(cn) * sigmoidf(z[3]);
        c_st[idx]  = cn;
        hl_bf[idx] = f2bf(hl);
        if (t == T - 1) out[OUT_CL + idx] = cn;
      }
    }
    gbar(bar_cnt, bar_gen);

    // ================= stage 2: backbone feat = lecun_tanh([x,h]@bbW.T) ====
    if (wave < 2) {
      int m0b = rb0 + 16 * wave;
      v8f acc = zero8();
#pragma unroll 4
      for (int k0 = 0; k0 < H; k0 += 32) {
        v16bf a  = load_a16(hl_bf, H, m0b, k0, lane);
        v16bf bm = load_b16_off(bbb, F + H, ct16, F, k0, lane);
        acc = __builtin_amdgcn_wmma_f32_16x16x32_bf16(
            false, a, false, bm, (short)0, acc, false, false);
      }
#pragma unroll
      for (int r = 0; r < 8; ++r) {
        int b = m0b + r + 8 * lhi;
        const float* xp = x + ((size_t)b * T + t) * F;
        float pre = acc[r] + bbbias;
#pragma unroll
        for (int f = 0; f < F; ++f) pre += xp[f] * bbx[f];
        float ftv = 1.7159f * tanhf(0.666f * pre);
        ft_bf[b * BBU + clane] = f2bf(ftv);
      }
    }
    gbar(bar_cnt, bar_gen);

    // ================= stage 3: ff1/ff2/time_a/time_b + gate ===============
    {
      v8f a1 = zero8(), a2 = zero8(), a3 = zero8(), a4 = zero8();
#pragma unroll
      for (int k0 = 0; k0 < BBU; k0 += 32) {
        int kc = k0 >> 5;
        v16bf a  = load_a16(ft_bf, BBU, m0, k0, lane);
        v16bf bm0 = load_b16_lds(ldsFF, (0 * 4 + ct) * 4 + kc, lane);
        v16bf bm1 = load_b16_lds(ldsFF, (1 * 4 + ct) * 4 + kc, lane);
        v16bf bm2 = load_b16_lds(ldsFF, (2 * 4 + ct) * 4 + kc, lane);
        v16bf bm3 = load_b16_lds(ldsFF, (3 * 4 + ct) * 4 + kc, lane);
        a1 = __builtin_amdgcn_wmma_f32_16x16x32_bf16(
            false, a, false, bm0, (short)0, a1, false, false);
        a2 = __builtin_amdgcn_wmma_f32_16x16x32_bf16(
            false, a, false, bm1, (short)0, a2, false, false);
        a3 = __builtin_amdgcn_wmma_f32_16x16x32_bf16(
            false, a, false, bm2, (short)0, a3, false, false);
        a4 = __builtin_amdgcn_wmma_f32_16x16x32_bf16(
            false, a, false, bm3, (short)0, a4, false, false);
      }
#pragma unroll
      for (int r = 0; r < 8; ++r) {
        int b = m0 + r + 8 * lhi;
        float f1 = tanhf(a1[r] + b_f1);
        float f2 = tanhf(a2[r] + b_f2);
        float ti = sigmoidf((a3[r] + b_ta) + (a4[r] + b_tb));
        float hn = f1 * (1.0f - ti) + ti * f2;
        int idx = b * H + jlane;
        h_bf[idx]  = f2bf(hn);
        h_f32[idx] = hn;
        if (t == T - 1) out[OUT_HL + idx] = hn;
      }
    }
    gbar(bar_cnt, bar_gen);

    // ================= stage 4: output heads (y_pred, e_pred) ==============
    {
      int g = threadIdx.x >> 4;            // 16-lane group
      if (g < 12) {                        // 4 rows x 3 outputs
        int row = g & 3;
        int o   = g >> 2;                  // 0,1 -> y ; 2 -> e
        int b   = w * 4 + row;
        int sub = threadIdx.x & 15;
        const float* hv = h_f32 + (size_t)b * H;
        const float* wv = (o < 2) ? (pred_W + o * H) : en_W;
        float s = 0.0f;
#pragma unroll 4
        for (int k = sub; k < H; k += 16) s += hv[k] * wv[k];
        s += __shfl_xor(s, 1, 16);
        s += __shfl_xor(s, 2, 16);
        s += __shfl_xor(s, 4, 16);
        s += __shfl_xor(s, 8, 16);
        if (sub == 0) {
          if (o < 2) out[OUT_Y + ((size_t)b * T + t) * 2 + o] = s + pred_b[o];
          else       out[OUT_E + (size_t)b * T + t]           = s + en_b[0];
        }
      }
    }
    // program order guarantees stage-4 reads of h_f32[t] finish before any
    // block can reach stage-3 of step t+1 (two barriers in between).
  }
}

// ============================================================================
extern "C" void kernel_launch(void* const* d_in, const int* in_sizes, int n_in,
                              void* d_out, int out_size, void* d_ws, size_t ws_size,
                              hipStream_t stream) {
  (void)in_sizes; (void)n_in; (void)out_size; (void)ws_size;
  const float* x       = (const float*)d_in[0];
  const float* lstm_Wi = (const float*)d_in[1];
  const float* lstm_bi = (const float*)d_in[2];
  const float* lstm_Wh = (const float*)d_in[3];
  const float* bb_W    = (const float*)d_in[4];
  const float* bb_b    = (const float*)d_in[5];
  const float* ff1_W   = (const float*)d_in[6];
  const float* ff1_b   = (const float*)d_in[7];
  const float* ff2_W   = (const float*)d_in[8];
  const float* ff2_b   = (const float*)d_in[9];
  const float* ta_W    = (const float*)d_in[10];
  const float* ta_b    = (const float*)d_in[11];
  const float* tb_W    = (const float*)d_in[12];
  const float* tb_b    = (const float*)d_in[13];
  const float* pred_W  = (const float*)d_in[14];
  const float* pred_b  = (const float*)d_in[15];
  const float* en_W    = (const float*)d_in[16];
  const float* en_b    = (const float*)d_in[17];

  unsigned char* ws = (unsigned char*)d_ws;
  float* out = (float*)d_out;

  // allow the full 320 KB per-WGP dynamic LDS allocation (idempotent, host-side)
  static bool attr_set = [] {
    (void)hipFuncSetAttribute((const void*)eblnn_persistent,
                              hipFuncAttributeMaxDynamicSharedMemorySize,
                              (int)LDS_BYTES);
    return true;
  }();
  (void)attr_set;

  eblnn_init<<<256, 256, 0, stream>>>(lstm_Wh, bb_W, ff1_W, ff2_W, ta_W, tb_W, ws);

  eblnn_persistent<<<NBLK, NTHR, LDS_BYTES, stream>>>(
      x, lstm_Wi, lstm_bi, bb_W, bb_b,
      ff1_b, ff2_b, ta_b, tb_b,
      pred_W, pred_b, en_W, en_b,
      ws, out);
}
